// SpatialAlignTransnormer_65841848648190
// MI455X (gfx1250) — matrix operand order repaired and checked by reading, not compiled, not run on hardware
//
#include <hip/hip_runtime.h>
#include <hip/hip_bf16.h>
#include <math.h>

// ---------------- problem constants (idx == 10 path) ----------------
#define B_      8
#define C1_     256
#define C2_     512
#define HH      56
#define LL      3136          // 56*56
#define MM      25088         // B_*LL
#define HD_     32            // head dim
#define A3SA_N  25690112      // 8*256*112*112
#define A4UP_N  51380224      // 8*512*112*112

typedef __attribute__((ext_vector_type(8)))  float    v8f;
typedef __attribute__((ext_vector_type(16))) _Float16 v16h;

union FragH { v16h v; _Float16 h[16]; };

__device__ __forceinline__ v8f zero_v8f() {
  v8f z = {0.f,0.f,0.f,0.f,0.f,0.f,0.f,0.f};
  return z;
}

// 2D sincos pos embedding: l = gw*56 + gh (meshgrid 'ij' flatten), dim 256
__device__ __forceinline__ float pos_val(int l, int c) {
  int gw = l / HH, gh = l % HH;
  int grp = c >> 6, k = c & 63;
  float omega = __expf(-(float)k * (9.210340371976184f / 64.f)); // 10000^(-k/64)
  float ang = ((grp < 2) ? (float)gw : (float)gh) * omega;
  return ((grp & 1) == 0) ? __sinf(ang) : __cosf(ang);
}

// block(256) = 8 waves of 32; full block sum
__device__ __forceinline__ float blk_sum(float v, float* sh) {
  int lane = threadIdx.x & 31, w = threadIdx.x >> 5;
  #pragma unroll
  for (int o = 16; o > 0; o >>= 1) v += __shfl_xor(v, o, 32);
  if (lane == 0) sh[w] = v;
  __syncthreads();
  float t = (threadIdx.x < 8) ? sh[threadIdx.x] : 0.f;
  if (w == 0) {
    #pragma unroll
    for (int o = 4; o > 0; o >>= 1) t += __shfl_xor(t, o, 32);
    if (lane == 0) sh[0] = t;
  }
  __syncthreads();
  float r = sh[0];
  __syncthreads();
  return r;
}

// ---------------- WMMA GEMM:  C[M,N] = A[M,K] @ Wt[N,K]^T + bias ----------------
// block tile 128x64, 8 waves (4x2), wave tile 32x32 = 2x2 v_wmma_f32_16x16x32_f16
__global__ __launch_bounds__(256)
void k_gemm_wmma(const float* __restrict__ A, const float* __restrict__ Wt,
                 const float* __restrict__ bias, float* __restrict__ C,
                 int M, int N, int K) {
  __shared__ _Float16 Ash[128][40];   // padded stride, 16B-aligned sub-chunks
  __shared__ _Float16 Bsh[64][40];
  const int tid  = threadIdx.x;
  const int wid  = tid >> 5, lane = tid & 31;
  const int wm   = wid & 3,  wn   = wid >> 2;
  const int g    = lane >> 4, hl  = lane & 15;
  const int m0   = blockIdx.x * 128;
  const int n0   = blockIdx.y * 64;

  v8f acc[2][2];
  acc[0][0] = zero_v8f(); acc[0][1] = zero_v8f();
  acc[1][0] = zero_v8f(); acc[1][1] = zero_v8f();

  for (int kk = 0; kk < K; kk += 32) {
    // stage A tile 128x32 (f32 -> f16), 16 elems/thread
    #pragma unroll
    for (int i = 0; i < 4; ++i) {
      int idx = (tid + 256 * i) << 2;
      int r = idx >> 5, c = idx & 31;
      const float4 f = *(const float4*)(A + (size_t)(m0 + r) * K + kk + c);
      Ash[r][c+0] = (_Float16)f.x; Ash[r][c+1] = (_Float16)f.y;
      Ash[r][c+2] = (_Float16)f.z; Ash[r][c+3] = (_Float16)f.w;
    }
    // stage B tile 64x32 from Wt[N,K]
    #pragma unroll
    for (int i = 0; i < 2; ++i) {
      int idx = (tid + 256 * i) << 2;
      int r = idx >> 5, c = idx & 31;
      const float4 f = *(const float4*)(Wt + (size_t)(n0 + r) * K + kk + c);
      Bsh[r][c+0] = (_Float16)f.x; Bsh[r][c+1] = (_Float16)f.y;
      Bsh[r][c+2] = (_Float16)f.z; Bsh[r][c+3] = (_Float16)f.w;
    }
    __syncthreads();

    FragH fa[2], fb[2];
    #pragma unroll
    for (int tm = 0; tm < 2; ++tm) {
      // A 16x32 f16 layout: lane g*16+m; element e -> k = (e/8)*16 + g*8 + (e%8)
      const _Float16* p = &Ash[wm * 32 + tm * 16 + hl][g * 8];
      #pragma unroll
      for (int e = 0; e < 8; ++e) { fa[tm].h[e] = p[e]; fa[tm].h[8 + e] = p[16 + e]; }
    }
    #pragma unroll
    for (int tn = 0; tn < 2; ++tn) {
      // B 32x16 f16 layout: lane g*16+n; element e -> k = (e/8)*16 + g*8 + (e%8)
      const _Float16* p = &Bsh[wn * 32 + tn * 16 + hl][g * 8];
      #pragma unroll
      for (int e = 0; e < 8; ++e) { fb[tn].h[e] = p[e]; fb[tn].h[8 + e] = p[16 + e]; }
    }
    #pragma unroll
    for (int tm = 0; tm < 2; ++tm)
      #pragma unroll
      for (int tn = 0; tn < 2; ++tn)
        acc[tm][tn] = __builtin_amdgcn_wmma_f32_16x16x32_f16(
            false, fa[tm].v, false, fb[tn].v, (short)0, acc[tm][tn], false, false);
    __syncthreads();
  }

  // C layout: lane = (m/8)*16 + n; vgpr r -> m = 8*g + r
  #pragma unroll
  for (int tm = 0; tm < 2; ++tm) {
    #pragma unroll
    for (int tn = 0; tn < 2; ++tn) {
      int gm0 = m0 + wm * 32 + tm * 16 + 8 * g;
      int gn  = n0 + wn * 32 + tn * 16 + hl;
      float bv = bias[gn];
      #pragma unroll
      for (int r = 0; r < 8; ++r)
        C[(size_t)(gm0 + r) * N + gn] = acc[tm][tn][r] + bv;
    }
  }
}

// ---------------- fused elementwise kernels ----------------

// a3 [8,256,112,112] -> 2x2 avg pool -> token-major X3[t,c]; X3p = X3 + pos
__global__ __launch_bounds__(256)
void k_avgpool_pos(const float* __restrict__ a3, float* __restrict__ X3,
                   float* __restrict__ X3p) {
  int t = blockIdx.x, c = threadIdx.x;
  int b = t / LL, l = t % LL, y = l / HH, x = l % HH;
  const float* p = a3 + ((size_t)((b * C1_ + c) * 112 + 2 * y)) * 112 + 2 * x;
  float v = 0.25f * (p[0] + p[1] + p[112] + p[113]);
  X3 [(size_t)t * C1_ + c] = v;
  X3p[(size_t)t * C1_ + c] = v + pos_val(l, c);
}

// a4 [8,512,56,56] -> token-major [25088, 512]
__global__ __launch_bounds__(256)
void k_a4_tokens(const float* __restrict__ a4, float* __restrict__ T4) {
  int t = blockIdx.x;
  int b = t / LL, l = t % LL, y = l / HH, x = l % HH;
  #pragma unroll
  for (int i = 0; i < 2; ++i) {
    int c = threadIdx.x + 256 * i;
    T4[(size_t)t * C2_ + c] = a4[((size_t)(b * C2_ + c) * HH + y) * HH + x];
  }
}

// in-place: BatchNorm (running stats 0/1, eval) + pos embedding
__global__ __launch_bounds__(256)
void k_bn_pos(float* __restrict__ X, const float* __restrict__ w,
              const float* __restrict__ bb) {
  int t = blockIdx.x, c = threadIdx.x, l = t % LL;
  size_t o = (size_t)t * C1_ + c;
  float v = X[o];
  v = v * 0.9999950000374997f * w[c] + bb[c];   // 1/sqrt(1+1e-5)
  X[o] = v + pos_val(l, c);
}

// in-place: LayerNorm(256) then L2-normalize(256) with eps clamp
__global__ __launch_bounds__(256)
void k_ln_l2(float* __restrict__ X, const float* __restrict__ w,
             const float* __restrict__ bb) {
  __shared__ float sh[8];
  size_t t = blockIdx.x; int c = threadIdx.x;
  size_t o = t * C1_ + c;
  float v = X[o];
  float mu  = blk_sum(v, sh) * (1.f / 256.f);
  float d   = v - mu;
  float var = blk_sum(d * d, sh) * (1.f / 256.f);
  float y   = d * rsqrtf(var + 1e-5f) * w[c] + bb[c];
  float n2  = blk_sum(y * y, sh);
  X[o] = y / fmaxf(sqrtf(n2), 1e-4f);
}

// OUT = RES + LayerNorm(X)
__global__ __launch_bounds__(256)
void k_ln_res(const float* __restrict__ X, const float* __restrict__ RES,
              const float* __restrict__ w, const float* __restrict__ bb,
              float* __restrict__ OUT) {
  __shared__ float sh[8];
  size_t t = blockIdx.x; int c = threadIdx.x;
  size_t o = t * C1_ + c;
  float v = X[o];
  float mu  = blk_sum(v, sh) * (1.f / 256.f);
  float d   = v - mu;
  float var = blk_sum(d * d, sh) * (1.f / 256.f);
  OUT[o] = RES[o] + d * rsqrtf(var + 1e-5f) * w[c] + bb[c];
}

// exact GELU, in place
__global__ __launch_bounds__(256)
void k_gelu(float* __restrict__ X) {
  size_t i = (size_t)blockIdx.x * 256 + threadIdx.x;
  float x = X[i];
  X[i] = 0.5f * x * (1.f + erff(x * 0.70710678118654752f));
}

// cosFormer: KV[n, i, j] = sum_l reweight(K)[l,i] * V[l,j], n = b*8+head, i<64, j<32
__global__ __launch_bounds__(256)
void k_kv(const float* __restrict__ Kb, const float* __restrict__ Vb,
          float* __restrict__ KV) {
  int n = blockIdx.x, b = n >> 3, hd = n & 7;
  int j = threadIdx.x & 31, ig = threadIdx.x >> 5;   // ig*8 .. ig*8+7 rows of KV
  __shared__ float Ks[8][32], Vs[8][32];
  float acc[8] = {0.f,0.f,0.f,0.f,0.f,0.f,0.f,0.f};
  for (int l0 = 0; l0 < LL; l0 += 8) {
    size_t base = ((size_t)(b * LL + l0 + ig)) * C1_ + hd * HD_ + j;
    Ks[ig][j] = Kb[base];
    Vs[ig][j] = Vb[base];
    __syncthreads();
    #pragma unroll
    for (int u = 0; u < 8; ++u) {
      float wv = (float)(l0 + u + 1) * (1.f / (float)LL);
      float cw = __cosf(wv), sw = __sinf(wv);
      float vj = Vs[u][j];
      #pragma unroll
      for (int r = 0; r < 8; ++r) {
        int i = ig * 8 + r;
        float kf = (i < 32) ? cw * Ks[u][i] : sw * Ks[u][i - 32];
        acc[r] += kf * vj;
      }
    }
    __syncthreads();
  }
  #pragma unroll
  for (int r = 0; r < 8; ++r)
    KV[((size_t)n * 64 + ig * 8 + r) * 32 + j] = acc[r];
}

// attn[l, head, j] = RMSNorm_j( Q_(l) @ KV );  one wave per token (32 lanes = head dim)
__global__ __launch_bounds__(256)
void k_attn(const float* __restrict__ Qb, const float* __restrict__ KV,
            const float* __restrict__ rms_w, float* __restrict__ ATT) {
  int n = blockIdx.x, b = n >> 3, hd = n & 7;
  int j = threadIdx.x & 31, lt = threadIdx.x >> 5;
  __shared__ float KVs[64][32];
  __shared__ float Qs[8][32];
  #pragma unroll
  for (int i = 0; i < 8; ++i) {
    int e = threadIdx.x + 256 * i;
    KVs[e >> 5][e & 31] = KV[(size_t)n * 2048 + e];
  }
  int l = blockIdx.y * 8 + lt;
  size_t qbase = ((size_t)(b * LL + l)) * C1_ + hd * HD_;
  Qs[lt][j] = Qb[qbase + j];
  __syncthreads();
  float wv = (float)(l + 1) * (1.f / (float)LL);
  float cw = __cosf(wv), sw = __sinf(wv);
  float s = 0.f;
  #pragma unroll
  for (int i = 0; i < 32; ++i) {
    float q = Qs[lt][i];
    s += q * (cw * KVs[i][j] + sw * KVs[32 + i][j]);
  }
  float ss = s * s;
  #pragma unroll
  for (int o = 16; o > 0; o >>= 1) ss += __shfl_xor(ss, o, 32);
  float rms = sqrtf(ss * (1.f / 32.f) + 1e-6f);
  ATT[qbase + j] = rms_w[j] * s / rms;
}

// ---------------- bilinear x2 upsample (half-pixel) ----------------

__global__ __launch_bounds__(256)
void k_a3sa(const float* __restrict__ OUT2, const float* __restrict__ a3,
            float* __restrict__ out) {
  size_t idx = (size_t)blockIdx.x * 256 + threadIdx.x;       // < A3SA_N
  int ox = (int)(idx % 112);
  int oy = (int)((idx / 112) % 112);
  int ch = (int)((idx / 12544) % C1_);
  int b  = (int)(idx / ((size_t)12544 * C1_));
  float sx = ox * 0.5f - 0.25f, sy = oy * 0.5f - 0.25f;
  int x0 = (int)floorf(sx), y0 = (int)floorf(sy);
  float fx = sx - x0, fy = sy - y0;
  int x0c = max(x0, 0), x1c = min(x0 + 1, HH - 1);
  int y0c = max(y0, 0), y1c = min(y0 + 1, HH - 1);
  size_t tb = (size_t)b * LL;
  float v00 = OUT2[(tb + y0c * HH + x0c) * C1_ + ch];
  float v01 = OUT2[(tb + y0c * HH + x1c) * C1_ + ch];
  float v10 = OUT2[(tb + y1c * HH + x0c) * C1_ + ch];
  float v11 = OUT2[(tb + y1c * HH + x1c) * C1_ + ch];
  float v = (1.f - fy) * ((1.f - fx) * v00 + fx * v01)
          +        fy  * ((1.f - fx) * v10 + fx * v11);
  out[idx] = v * a3[idx];
}

__global__ __launch_bounds__(256)
void k_a4up(const float* __restrict__ a4, float* __restrict__ out) {
  size_t idx = (size_t)blockIdx.x * 256 + threadIdx.x;       // < A4UP_N
  int ox = (int)(idx % 112);
  int oy = (int)((idx / 112) % 112);
  int ch = (int)((idx / 12544) % C2_);
  int b  = (int)(idx / ((size_t)12544 * C2_));
  float sx = ox * 0.5f - 0.25f, sy = oy * 0.5f - 0.25f;
  int x0 = (int)floorf(sx), y0 = (int)floorf(sy);
  float fx = sx - x0, fy = sy - y0;
  int x0c = max(x0, 0), x1c = min(x0 + 1, HH - 1);
  int y0c = max(y0, 0), y1c = min(y0 + 1, HH - 1);
  const float* p = a4 + ((size_t)(b * C2_ + ch)) * LL;
  float v00 = p[y0c * HH + x0c], v01 = p[y0c * HH + x1c];
  float v10 = p[y1c * HH + x0c], v11 = p[y1c * HH + x1c];
  out[idx] = (1.f - fy) * ((1.f - fx) * v00 + fx * v01)
           +        fy  * ((1.f - fx) * v10 + fx * v11);
}

// ---------------- launcher ----------------
extern "C" void kernel_launch(void* const* d_in, const int* in_sizes, int n_in,
                              void* d_out, int out_size, void* d_ws, size_t ws_size,
                              hipStream_t stream) {
  (void)in_sizes; (void)n_in; (void)out_size; (void)ws_size;
  const float* a3     = (const float*)d_in[0];
  const float* a4     = (const float*)d_in[1];
  const float* conv_w = (const float*)d_in[2];
  const float* conv_b = (const float*)d_in[3];
  const float* bn_w   = (const float*)d_in[4];
  const float* bn_b   = (const float*)d_in[5];
  const float* q_w    = (const float*)d_in[6];
  const float* q_b    = (const float*)d_in[7];
  const float* q_ln_w = (const float*)d_in[8];
  const float* q_ln_b = (const float*)d_in[9];
  const float* k_w    = (const float*)d_in[10];
  const float* k_b    = (const float*)d_in[11];
  const float* k_ln_w = (const float*)d_in[12];
  const float* k_ln_b = (const float*)d_in[13];
  const float* v_w    = (const float*)d_in[14];
  const float* v_b    = (const float*)d_in[15];
  const float* o_w    = (const float*)d_in[16];
  const float* o_b    = (const float*)d_in[17];
  const float* rms_w  = (const float*)d_in[18];
  const float* l1_w   = (const float*)d_in[19];
  const float* l1_b   = (const float*)d_in[20];
  const float* l2_w   = (const float*)d_in[21];
  const float* l2_b   = (const float*)d_in[22];
  const float* n1_w   = (const float*)d_in[23];
  const float* n1_b   = (const float*)d_in[24];
  const float* n2_w   = (const float*)d_in[25];
  const float* n2_b   = (const float*)d_in[26];
  // d_in[27] == idx == 10 (reference path baked in)

  float* ws = (float*)d_ws;
  const size_t SZ = (size_t)MM * C1_;        // 6,422,528 floats per [M,256] buffer
  float* X3   = ws;                          // residual tokens (live to first ln_res)
  float* X3p  = ws + 1 * SZ;                 // a3 tokens + pos; later reused as ATT
  float* A4F  = ws + 2 * SZ;                 // conv+BN+pos tokens; later reused as OUT1
  float* Qb   = ws + 3 * SZ;                 // Q; later reused as FFN2 temp
  float* Kb   = ws + 4 * SZ;                 // K; later reused as OUT2
  float* Vb   = ws + 5 * SZ;                 // V; later reused as O-proj temp
  float* KVb  = ws + 6 * SZ;                 // [64,64,32] = 131072 floats
  float* BIG  = ws + 6 * SZ + 262144;        // a4 tokens [M,512], then FFN hidden [M,1024]
  float* ATT  = X3p;
  float* OUT1 = A4F;
  float* OUT2 = Kb;
  float* outp = (float*)d_out;

  k_avgpool_pos<<<MM, 256, 0, stream>>>(a3, X3, X3p);
  k_a4_tokens  <<<MM, 256, 0, stream>>>(a4, BIG);
  // conv 1x1 as GEMM  [M,512]x[512,256]
  k_gemm_wmma<<<dim3(MM / 128, C1_ / 64), 256, 0, stream>>>(BIG, conv_w, conv_b, A4F, MM, C1_, C2_);
  k_bn_pos<<<MM, 256, 0, stream>>>(A4F, bn_w, bn_b);
  // Q / K / V projections
  k_gemm_wmma<<<dim3(MM / 128, C1_ / 64), 256, 0, stream>>>(X3p, q_w, q_b, Qb, MM, C1_, C1_);
  k_ln_l2<<<MM, 256, 0, stream>>>(Qb, q_ln_w, q_ln_b);
  k_gemm_wmma<<<dim3(MM / 128, C1_ / 64), 256, 0, stream>>>(A4F, k_w, k_b, Kb, MM, C1_, C1_);
  k_ln_l2<<<MM, 256, 0, stream>>>(Kb, k_ln_w, k_ln_b);
  k_gemm_wmma<<<dim3(MM / 128, C1_ / 64), 256, 0, stream>>>(A4F, v_w, v_b, Vb, MM, C1_, C1_);
  // cosFormer linear attention + fused RMSNorm
  k_kv  <<<64, 256, 0, stream>>>(Kb, Vb, KVb);
  k_attn<<<dim3(64, LL / 8), 256, 0, stream>>>(Qb, KVb, rms_w, ATT);
  // output projection + residual/LN
  k_gemm_wmma<<<dim3(MM / 128, C1_ / 64), 256, 0, stream>>>(ATT, o_w, o_b, Vb, MM, C1_, C1_);
  k_ln_res<<<MM, 256, 0, stream>>>(Vb, X3, n1_w, n1_b, OUT1);
  // FFN
  k_gemm_wmma<<<dim3(MM / 128, 1024 / 64), 256, 0, stream>>>(OUT1, l1_w, l1_b, BIG, MM, 1024, C1_);
  k_gelu<<<(MM * 1024) / 256, 256, 0, stream>>>(BIG);
  k_gemm_wmma<<<dim3(MM / 128, C1_ / 64), 256, 0, stream>>>(BIG, l2_w, l2_b, Qb, MM, C1_, 1024);
  k_ln_res<<<MM, 256, 0, stream>>>(Qb, OUT1, n2_w, n2_b, OUT2);
  // outputs
  k_a3sa<<<A3SA_N / 256, 256, 0, stream>>>(OUT2, a3, outp);
  k_a4up<<<A4UP_N / 256, 256, 0, stream>>>(a4, outp + A3SA_N);
}
